// ConvBlock_33509334844061
// MI455X (gfx1250) — matrix operand order repaired and checked by reading, not compiled, not run on hardware
//
#include <hip/hip_runtime.h>

// ---------------- types ----------------
typedef __attribute__((ext_vector_type(16))) __bf16 v16bf;
typedef __attribute__((ext_vector_type(8)))  float  v8f;

#define OUT_C   128
#define IN_C    64
#define KSZ     3
#define NB      64            // batch
#define HW      64            // H == W
#define KDIM    576           // IN_C*K*K
#define KCHUNKS 18            // 576 / 32
#define NHW_F   262144.0f     // N*H*W per channel
#define BN_EPS  1e-5f

// A-pack: [chunk(18)][mtile(8)][lane(32)][16] bf16 per plane
#define APACK_ELEMS (KCHUNKS * 8 * 32 * 16)   // 73728

// ---------------- bf16 helpers (explicit RNE, portable) ----------------
static __device__ __forceinline__ __bf16 f2bf(float f) {
    unsigned u = __float_as_uint(f);
    unsigned r = (u + 0x7FFFu + ((u >> 16) & 1u)) >> 16;
    union { unsigned short s; __bf16 b; } cv;
    cv.s = (unsigned short)r;
    return cv.b;
}
static __device__ __forceinline__ float bf2f(__bf16 b) {
    union { __bf16 b; unsigned short s; } cv;
    cv.b = b;
    return __uint_as_float(((unsigned)cv.s) << 16);
}

// ---------------- kernel 0: zero BN accumulators ----------------
__global__ __launch_bounds__(256) void zero_acc(float* acc) {
    acc[threadIdx.x] = 0.0f;   // 256 floats: [0..127]=sum, [128..255]=sumsq
}

// ---------------- kernel 1: pre-pack block weights into WMMA A fragments ----------------
// W2d[co, j] = weight[co>>3][j>>3][co&7][j&7],  j = ci*9 + kh*3 + kw  (j < 576)
// A fragment layout packed here (16-bit A 16x32): lane<16 holds row M=lane, elems = K 0..15;
// lane>=16 holds row M=lane-16, elems = K 16..31.
__global__ __launch_bounds__(256) void prepack_w(const float* __restrict__ weight,
                                                 __bf16* __restrict__ Ahi,
                                                 __bf16* __restrict__ Alo) {
    int i = blockIdx.x * 256 + threadIdx.x;          // 0 .. 73727
    if (i >= APACK_ELEMS) return;
    int el   = i & 15;
    int lane = (i >> 4) & 31;
    int mt   = (i >> 9) & 7;
    int kc   = i >> 12;
    int co   = mt * 16 + (lane & 15);
    int kl   = el + ((lane >> 4) << 4);
    int j    = kc * 32 + kl;                         // always < 576
    float v  = weight[((co >> 3) * 72 + (j >> 3)) * 64 + ((co & 7) << 3) + (j & 7)];
    __bf16 hi = f2bf(v);
    __bf16 lo = f2bf(v - bf2f(hi));
    Ahi[i] = hi;
    Alo[i] = lo;
}

// ---------------- kernel 2: implicit-GEMM conv via compensated bf16 WMMA ----------------
// Grid: 2048 blocks of 256 threads (8 waves). Block tile: M=128 x N=128 spatial.
// Double-buffered LDS staging: loads for chunk kc+1 are issued before the 24 WMMAs of
// chunk kc so the XDL work hides global-load latency; convert+store after compute.
__global__ __launch_bounds__(256) void conv_wmma(const float* __restrict__ x,
                                                 const __bf16* __restrict__ Ahi,
                                                 const __bf16* __restrict__ Alo,
                                                 float* __restrict__ out,
                                                 float* __restrict__ bnAcc) {
    // B-fragment staging, 2 buffers: [buf][ntile(8)][lane(32)][16] bf16 (32 KB total)
    __shared__ __align__(32) __bf16 lds_hi[2][8 * 32 * 16];
    __shared__ __align__(32) __bf16 lds_lo[2][8 * 32 * 16];

    const int tid    = threadIdx.x;
    const int wave   = tid >> 5;
    const int laneid = tid & 31;

    const int s0  = blockIdx.x * 128;      // first spatial column of tile
    const int n   = s0 >> 12;              // batch index (4096 spatial per image)
    const int hw0 = s0 & 4095;

    // ---- per-thread staging constants ----
    const int c  = tid & 127;              // spatial column owned by this thread
    const int b  = tid >> 7;               // 0/1: low/high K parity
    const int hw = hw0 + c;
    const int hb = hw >> 6;
    const int wb = hw & 63;
    const int xbase = ((n * IN_C) * HW + hb) * HW + wb;   // x[n,0,hb,wb]
    // 9-bit halo validity mask over r9 = kh*3+kw
    int vmask = 0;
#pragma unroll
    for (int r9c = 0; r9c < 9; ++r9c) {
        int kh = r9c / 3, kw = r9c % 3;
        int ok = ((unsigned)(hb + kh - 1) < 64u) && ((unsigned)(wb + kw - 1) < 64u);
        vmask |= ok << r9c;
    }
    // LDS store base: u = c>>4, lane = (c&15) (+16 for e>=8), el = 2*(e&7)+b
    const int sbase = (((c >> 4) * 32 + (c & 15)) * 16) + b;

    const v16bf* __restrict__ Ahi_v = (const v16bf*)Ahi;
    const v16bf* __restrict__ Alo_v = (const v16bf*)Alo;

    v8f acc[8];
#pragma unroll
    for (int u = 0; u < 8; ++u) acc[u] = (v8f){0.f,0.f,0.f,0.f,0.f,0.f,0.f,0.f};

    // incremental K-index state: j = (step count)*2 + b, j = ci*9 + r9
    int ci = 0, r9 = b;
    float vreg[16];

    // ---- stage loads for one chunk: 16 predicated global loads, j advances by +2 each ----
#define STAGE_LOADS()                                                          \
    _Pragma("unroll")                                                          \
    for (int e = 0; e < 16; ++e) {                                             \
        int kh   = (r9 >= 3) + (r9 >= 6);                                      \
        int kw   = r9 - kh * 3;                                                \
        int addr = xbase + (ci << 12) + (kh << 6) + kw - 65;                   \
        vreg[e]  = ((vmask >> r9) & 1) ? x[addr] : 0.0f;                       \
        r9 += 2;                                                               \
        int wrap = (r9 >= 9);                                                  \
        ci += wrap;                                                            \
        r9 -= wrap ? 9 : 0;                                                    \
    }

    // ---- convert + store staged values into LDS fragment layout ----
#define STAGE_STORE(buf)                                                       \
    _Pragma("unroll")                                                          \
    for (int e = 0; e < 16; ++e) {                                             \
        float v   = vreg[e];                                                   \
        __bf16 hi = f2bf(v);                                                   \
        __bf16 lo = f2bf(v - bf2f(hi));                                        \
        int off   = sbase + ((e >= 8) ? 256 : 0) + 2 * (e & 7);                \
        lds_hi[buf][off] = hi;                                                 \
        lds_lo[buf][off] = lo;                                                 \
    }

    // prologue: chunk 0 -> buffer 0
    STAGE_LOADS();
    STAGE_STORE(0);
    __syncthreads();

    for (int kc = 0; kc < KCHUNKS; ++kc) {
        const int cur = kc & 1;
        // issue next chunk's global loads first (latency hidden by WMMAs below)
        if (kc + 1 < KCHUNKS) { STAGE_LOADS(); }

        // A fragments for this wave's 16 output channels (L2-resident pack)
        int abase = (kc * 8 + wave) * 32 + laneid;
        v16bf ah = Ahi_v[abase];
        v16bf al = Alo_v[abase];
        if (kc + 1 < KCHUNKS)   // gfx1250 global_prefetch_b8 for next chunk's fragments
            __builtin_prefetch(&Ahi_v[((kc + 1) * 8 + wave) * 32 + laneid], 0, 1);

        // 8 N-tiles x 3 compensated bf16 WMMAs
#pragma unroll
        for (int u = 0; u < 8; ++u) {
            v16bf bh = *(const v16bf*)&lds_hi[cur][(u * 32 + laneid) * 16];
            v16bf bl = *(const v16bf*)&lds_lo[cur][(u * 32 + laneid) * 16];
            acc[u] = __builtin_amdgcn_wmma_f32_16x16x32_bf16(false, ah, false, bh,
                                                             (short)0, acc[u], false, false);
            acc[u] = __builtin_amdgcn_wmma_f32_16x16x32_bf16(false, ah, false, bl,
                                                             (short)0, acc[u], false, false);
            acc[u] = __builtin_amdgcn_wmma_f32_16x16x32_bf16(false, al, false, bh,
                                                             (short)0, acc[u], false, false);
        }

        if (kc + 1 < KCHUNKS) { STAGE_STORE(!cur); }
        __syncthreads();
    }

    // ---- epilogue: write y, accumulate per-channel sum / sumsq ----
    // C/D layout: lane holds (M = r + 8*half, N = laneid&15) in VGPR r.
    const int half = laneid >> 4;              // 0 or 1
    const int ncol = laneid & 15;
#pragma unroll
    for (int r = 0; r < 8; ++r) {
        int co = wave * 16 + r + (half << 3);
        float ssum = 0.0f, ssq = 0.0f;
#pragma unroll
        for (int u = 0; u < 8; ++u) {
            float v = acc[u][r];
            ssum += v;
            ssq  += v * v;
            int hwo = hw0 + u * 16 + ncol;
            out[(n * OUT_C + co) * 4096 + hwo] = v;
        }
        // reduce the 16 lanes that share this channel (xor masks stay within the half)
#pragma unroll
        for (int m = 8; m >= 1; m >>= 1) {
            ssum += __shfl_xor(ssum, m, 32);
            ssq  += __shfl_xor(ssq,  m, 32);
        }
        if (ncol == 0) {
            atomicAdd(&bnAcc[co], ssum);
            atomicAdd(&bnAcc[OUT_C + co], ssq);
        }
    }
}

// ---------------- kernel 3: BN stats -> scale/shift ----------------
__global__ __launch_bounds__(128) void bn_finalize(const float* __restrict__ bnAcc,
                                                   const float* __restrict__ gamma,
                                                   const float* __restrict__ beta,
                                                   float* __restrict__ ss) {
    int ch = threadIdx.x;
    float mean = bnAcc[ch] * (1.0f / NHW_F);
    float var  = bnAcc[OUT_C + ch] * (1.0f / NHW_F) - mean * mean;
    float inv  = rsqrtf(var + BN_EPS);
    ss[ch]         = gamma[ch] * inv;
    ss[OUT_C + ch] = beta[ch] - gamma[ch] * mean * inv;
}

// ---------------- kernel 4: y = relu(y*scale + shift), vectorized ----------------
__global__ __launch_bounds__(256) void bn_apply(float4* __restrict__ out,
                                                const float* __restrict__ ss) {
    int i4 = blockIdx.x * 256 + threadIdx.x;       // 8,388,608 float4's
    int co = ((i4 << 2) >> 12) & (OUT_C - 1);
    float sc = ss[co], sh = ss[OUT_C + co];
    float4 v = out[i4];
    v.x = fmaxf(fmaf(v.x, sc, sh), 0.0f);
    v.y = fmaxf(fmaf(v.y, sc, sh), 0.0f);
    v.z = fmaxf(fmaf(v.z, sc, sh), 0.0f);
    v.w = fmaxf(fmaf(v.w, sc, sh), 0.0f);
    out[i4] = v;
}

// ---------------- host launcher ----------------
extern "C" void kernel_launch(void* const* d_in, const int* in_sizes, int n_in,
                              void* d_out, int out_size, void* d_ws, size_t ws_size,
                              hipStream_t stream) {
    const float* x      = (const float*)d_in[0];
    const float* weight = (const float*)d_in[1];
    const float* gamma  = (const float*)d_in[2];
    const float* beta   = (const float*)d_in[3];
    float* out = (float*)d_out;

    // workspace layout
    float*  bnAcc = (float*)d_ws;                                    // 256 floats
    float*  ss    = bnAcc + 256;                                     // 256 floats
    __bf16* Ahi   = (__bf16*)((char*)d_ws + 2048);                   // 73728 bf16
    __bf16* Alo   = Ahi + APACK_ELEMS;                               // 73728 bf16

    zero_acc<<<1, 256, 0, stream>>>(bnAcc);
    prepack_w<<<(APACK_ELEMS + 255) / 256, 256, 0, stream>>>(weight, Ahi, Alo);
    conv_wmma<<<2048, 256, 0, stream>>>(x, Ahi, Alo, out, bnAcc);
    bn_finalize<<<1, 128, 0, stream>>>(bnAcc, gamma, beta, ss);
    bn_apply<<<(NB * OUT_C * HW * HW / 4 + 255) / 256, 256, 0, stream>>>((float4*)out, ss);
}
